// StripMLP_Block_65180423684936
// MI455X (gfx1250) — matrix operand order repaired and checked by reading, not compiled, not run on hardware
//
#include <hip/hip_runtime.h>
#include <hip/hip_bf16.h>
#include <math.h>

// ---------------------------------------------------------------------------
// StripMLP block for MI455X (gfx1250, wave32).
// Big 1x1-conv GEMMs run on V_WMMA_F32_16X16X32_BF16 (bf16 in, f32 accum),
// activations staged through LDS with vectorized b128 loads; weights
// pre-packed into WMMA A-fragment lane layout. Strip-conv kernels use
// global_load_async_to_lds for their weight rows. Depthwise convs +
// attention are direct VALU kernels.
// ---------------------------------------------------------------------------

typedef __attribute__((ext_vector_type(16))) __bf16 v16bf;
typedef __attribute__((ext_vector_type(8)))  float  v8f;

#define HW   3136      // 56*56
#define NB   16        // batch
#define CCH  320
#define DH   160       // DIM_HALF
#define PTOT (NB*HW)   // 50176, divisible by 128

union frag16 { v16bf v; uint4 u[2]; };
union f4u    { float4 v; float f[4]; };
union bpack  { __bf16 h[2]; unsigned u32; };

__device__ __forceinline__ float gelu_exact(float x) {
    return 0.5f * x * (1.0f + erff(x * 0.70710678118654752f));
}

// Async raw copy of `nq` 16-byte quads from global to LDS (one quad per
// active lane), then drain ASYNCcnt. Emits global_load_async_to_lds_b128.
__device__ __forceinline__ void async_row_to_lds(const float* __restrict__ g,
                                                 float* __restrict__ lds,
                                                 int t, int nq) {
    if (t < nq) {
        unsigned long long ga = (unsigned long long)(g + t * 4);
        unsigned lo = (unsigned)(unsigned long long)(lds + t * 4);
        asm volatile("global_load_async_to_lds_b128 %0, %1, off"
                     :: "v"(lo), "v"(ga) : "memory");
    }
    asm volatile("s_wait_asynccnt 0x0" ::: "memory");
}

// ---------------------------------------------------------------------------
// Weight packing: fp32 [M,K] row-major -> bf16 WMMA A-fragments.
// Fragment (mt,kt): 32 lanes x 16 bf16 contiguous.
//   lane L<16 : M = mt*16+L, K = kt*32 + {0..7,16..23}
//   lane L>=16: M = mt*16+(L-16), K = kt*32 + {8..15,24..31}
// ---------------------------------------------------------------------------
__global__ __launch_bounds__(256) void pack_w_k(const float* __restrict__ W,
                                                __bf16* __restrict__ F,
                                                int M, int K) {
    int KT = K >> 5;
    int total = (M >> 4) * KT * 32 * 16;
    for (int idx = blockIdx.x * 256 + threadIdx.x; idx < total;
         idx += gridDim.x * 256) {
        int j    = idx & 15;
        int L    = (idx >> 4) & 31;
        int tile = idx >> 9;
        int kt   = tile % KT;
        int mt   = tile / KT;
        int m    = mt * 16 + (L & 15);
        int kk;
        if (L < 16) kk = (j < 8) ? j       : 16 + (j - 8);
        else        kk = (j < 8) ? 8 + j   : 24 + (j - 8);
        int k = kt * 32 + kk;
        F[idx] = (__bf16)W[(long)m * K + k];
    }
}

// BN folding: a = scale/sqrt(var+eps), b = bias - mean*a
__global__ __launch_bounds__(256) void bn_prep_k(const float* __restrict__ s,
                                                 const float* __restrict__ b,
                                                 const float* __restrict__ m,
                                                 const float* __restrict__ v,
                                                 float* __restrict__ oa,
                                                 float* __restrict__ ob, int n) {
    int i = blockIdx.x * 256 + threadIdx.x;
    if (i < n) {
        float inv = s[i] * rsqrtf(v[i] + 1e-5f);
        oa[i] = inv;
        ob[i] = b[i] - m[i] * inv;
    }
}

// pos[c][i*56+j] = table[(j-i+56)*160 + c]   (table is (113,160) row-major)
__global__ __launch_bounds__(256) void pos_prep_k(const float* __restrict__ tab,
                                                  float* __restrict__ pos) {
    int idx = blockIdx.x * 256 + threadIdx.x;
    if (idx >= DH * HW) return;
    int c = idx / HW, ij = idx - c * HW;
    int i = ij / 56, j = ij - i * 56;
    pos[idx] = tab[(j - i + 56) * DH + c];
}

// ---------------------------------------------------------------------------
// Tiled bf16 WMMA GEMM:  out[m,p] = epilogue( sum_k W[m,k] * X[k,p] )
// X is the channel-concat of two NCHW fp32 tensors (split at `split`,
// a multiple of 32 so every 32-wide k-chunk lies entirely in one source).
// Block: 256 threads = 8 waves, tile M=32 x P=128; each wave does 16x32.
// ---------------------------------------------------------------------------
__global__ __launch_bounds__(256) void gemm_bf16_k(
    const __bf16* __restrict__ Wfrag,
    const float*  __restrict__ X0, long ns0,
    const float*  __restrict__ X1, long ns1, int split,
    int K,
    const float* __restrict__ bias,
    const float* __restrict__ bn_a, const float* __restrict__ bn_b,
    int do_gelu,
    const float* __restrict__ posadd,
    float* __restrict__ out, long out_ns) {

    __shared__ __bf16 Xs[128 * 40];   // [p][k], row stride 40 (80B, 16B-aligned)

    const int tid  = threadIdx.x;
    const int lane = tid & 31;
    const int wave = tid >> 5;
    const int wrow = wave >> 2;           // 0..1 : which 16-row m sub-tile
    const int wcol = wave & 3;            // 0..3 : which 32-col p sub-tile
    const int p0   = blockIdx.x * 128;
    const int mblk = blockIdx.y;          // 32-row tile index
    const int mt   = mblk * 2 + wrow;     // 16-row tile index
    const int KT   = K >> 5;

    const int pl   = lane & 15;
    const int koff = (lane >> 4) * 8;

    // Per-thread staging geometry, hoisted out of the k-loop.
    // Unit i (i=0,1): k-pair base kk[i], p-quad pq[i]; fixed n/hw.
    int kk[2], pq[2];
    const float* sb0[2];
    const float* sb1[2];
    #pragma unroll
    for (int i = 0; i < 2; ++i) {
        int u  = i * 256 + tid;           // 0..511
        kk[i]  = (u >> 5) << 1;           // 0,2,...,30
        pq[i]  = (u & 31) << 2;           // 0,4,...,124
        int pp = p0 + pq[i];
        int n  = pp / HW;
        int hw = pp - n * HW;
        sb0[i] = X0 + (long)n * ns0 + hw;
        sb1[i] = (X1 ? X1 + (long)n * ns1 + hw : sb0[i]);
    }

    v8f acc0 = {}; v8f acc1 = {};

    for (int kt = 0; kt < KT; ++kt) {
        const int  k0 = kt << 5;
        const bool hi = (k0 >= split);    // whole chunk on one side of concat

        // ---- stage: 4 x global_load_b128, then convert/pack to LDS ----
        f4u r0a, r0b, r1a, r1b;
        {
            long c0 = hi ? (long)(k0 + kk[0] - split) : (long)(k0 + kk[0]);
            long c1 = hi ? (long)(k0 + kk[1] - split) : (long)(k0 + kk[1]);
            const float* s0 = (hi ? sb1[0] : sb0[0]) + c0 * HW;
            const float* s1 = (hi ? sb1[1] : sb0[1]) + c1 * HW;
            r0a.v = *(const float4*)s0;
            r0b.v = *(const float4*)(s0 + HW);
            r1a.v = *(const float4*)s1;
            r1b.v = *(const float4*)(s1 + HW);
        }
        #pragma unroll
        for (int j = 0; j < 4; ++j) {
            bpack w0; w0.h[0] = (__bf16)r0a.f[j]; w0.h[1] = (__bf16)r0b.f[j];
            *(unsigned*)&Xs[(pq[0] + j) * 40 + kk[0]] = w0.u32;
            bpack w1; w1.h[0] = (__bf16)r1a.f[j]; w1.h[1] = (__bf16)r1b.f[j];
            *(unsigned*)&Xs[(pq[1] + j) * 40 + kk[1]] = w1.u32;
        }
        __syncthreads();

        // ---- A fragment (pre-packed weights, L2-resident) ----
        frag16 afr;
        const uint4* ap =
            (const uint4*)(Wfrag + ((long)(mt * KT + kt) * 32 + lane) * 16);
        afr.u[0] = ap[0]; afr.u[1] = ap[1];
        if (kt + 1 < KT)
            __builtin_prefetch(Wfrag + ((long)(mt * KT + kt + 1) * 32 + lane) * 16,
                               0, 3);

        // ---- B fragments: two 16-byte LDS reads per lane (k = koff+0..7, koff+16..23) ----
        frag16 bfr0, bfr1;
        {
            const uint4* bp0 = (const uint4*)(&Xs[(wcol * 32 + pl) * 40 + koff]);
            bfr0.u[0] = bp0[0]; bfr0.u[1] = bp0[2];
            const uint4* bp1 =
                (const uint4*)(&Xs[(wcol * 32 + 16 + pl) * 40 + koff]);
            bfr1.u[0] = bp1[0]; bfr1.u[1] = bp1[2];
        }

        acc0 = __builtin_amdgcn_wmma_f32_16x16x32_bf16(
            false, afr.v, false, bfr0.v, (short)0, acc0, false, false);
        acc1 = __builtin_amdgcn_wmma_f32_16x16x32_bf16(
            false, afr.v, false, bfr1.v, (short)0, acc1, false, false);
        __syncthreads();
    }

    // Epilogue: C/D layout -> VGPR r: lanes 0-15 M=r, lanes 16-31 M=8+r; N=lane&15
    const int msub  = (lane >> 4) * 8;
    const int mbase = mblk * 32 + wrow * 16;
    #pragma unroll
    for (int half = 0; half < 2; ++half) {
        v8f a = half ? acc1 : acc0;
        int p  = p0 + wcol * 32 + half * 16 + pl;
        int n  = p / HW;
        int hw = p - n * HW;
        #pragma unroll
        for (int r = 0; r < 8; ++r) {
            int m = mbase + msub + r;
            float v = a[r];
            if (bias)    v += bias[m];
            if (bn_a)    v  = v * bn_a[m] + bn_b[m];
            if (do_gelu) v  = gelu_exact(v);
            if (posadd)  v += posadd[(long)m * HW + hw];
            out[(long)n * out_ns + (long)m * HW + hw] = v;
        }
    }
}

// ---------------------------------------------------------------------------
// proj_h: grouped strip conv along W. Input xh = x1 + pos_h read on the fly.
// ph[n][c1*80+a][b][w] = b[o] + sum_i sum_t xh[n][c1*80+(a*56+i)%80]
//                                           [(a*56+i)/80][w-1+t] * W[o,i,t]
// o = a*56 + b. grid(4480, 32=n*2+c1), block 64 (w).
// ---------------------------------------------------------------------------
__global__ __launch_bounds__(64) void proj_h_kernel(
    const float* __restrict__ xp, const float* __restrict__ posh,
    const float* __restrict__ W, const float* __restrict__ B,
    float* __restrict__ ph) {
    __shared__ float wl[168];
    int o  = blockIdx.x;
    int nc = blockIdx.y;
    int n  = nc >> 1, c1 = nc & 1;
    int t  = threadIdx.x;
    async_row_to_lds(W + (long)o * 168, wl, t, 42);   // 42 quads = 168 floats
    __syncthreads();
    if (t >= 56) return;
    int a = o / 56, b = o - a * 56;
    int w = t;
    float acc = B[o];
    for (int i = 0; i < 56; ++i) {
        int src = a * 56 + i;
        int cs  = src % 80;
        int hs  = src / 80;
        const float* xr = xp   + ((long)n * CCH + c1 * 80 + cs) * HW + hs * 56;
        const float* pr = posh + (long)(c1 * 80 + cs) * HW + hs * 56;
        #pragma unroll
        for (int tt = 0; tt < 3; ++tt) {
            int ww = w - 1 + tt;
            if (ww >= 0 && ww < 56)
                acc += (xr[ww] + pr[ww]) * wl[i * 3 + tt];
        }
    }
    ph[((long)n * DH + c1 * 80 + a) * HW + b * 56 + w] = acc;
}

// proj_w: same thing along H, acting on ah.
__global__ __launch_bounds__(64) void proj_w_kernel(
    const float* __restrict__ ah,
    const float* __restrict__ W, const float* __restrict__ B,
    float* __restrict__ pw) {
    __shared__ float wl[168];
    int o  = blockIdx.x;
    int nc = blockIdx.y;
    int n  = nc >> 1, c1 = nc & 1;
    int t  = threadIdx.x;
    async_row_to_lds(W + (long)o * 168, wl, t, 42);
    __syncthreads();
    if (t >= 56) return;
    int a = o / 56, b = o - a * 56;
    int h = t;
    float acc = B[o];
    for (int i = 0; i < 56; ++i) {
        int src = a * 56 + i;
        int cs  = src % 80;
        int ws  = src / 80;
        const float* col = ah + ((long)n * DH + c1 * 80 + cs) * HW + ws;
        #pragma unroll
        for (int tt = 0; tt < 3; ++tt) {
            int hh = h - 1 + tt;
            if (hh >= 0 && hh < 56)
                acc += col[hh * 56] * wl[i * 3 + tt];
        }
    }
    pw[((long)n * DH + c1 * 80 + a) * HW + h * 56 + b] = acc;
}

// ---------------------------------------------------------------------------
// Depthwise conv helpers (recomputed, never materialized).
// fc_h_w: 3x7 kernel (pad 1,3) -> "xw_path".  fc_w_w: 7x3 (pad 3,1) -> "xh_path".
// ---------------------------------------------------------------------------
__device__ __forceinline__ float dwconv(const float* __restrict__ x2,
                                        const float* __restrict__ w,
                                        int h, int wd, int KH, int KW) {
    int rh = KH >> 1, rw = KW >> 1;
    float s = 0.f;
    for (int dh = -rh; dh <= rh; ++dh) {
        int hh = h + dh;
        if (hh < 0 || hh >= 56) continue;
        for (int dw = -rw; dw <= rw; ++dw) {
            int ww = wd + dw;
            if (ww < 0 || ww >= 56) continue;
            s += x2[hh * 56 + ww] * w[(dh + rh) * KW + (dw + rw)];
        }
    }
    return s;
}

// Mean over HW of (xh_path + xw_path + x2). One block per (n,c).
__global__ __launch_bounds__(256) void att_reduce_k(
    const float* __restrict__ xp,
    const float* __restrict__ fch, const float* __restrict__ fcw,
    float* __restrict__ attmean) {
    __shared__ float wh[21], wv[21], red[256];
    int nc = blockIdx.x;
    int n = nc / DH, c = nc - n * DH;
    int t = threadIdx.x;
    if (t < 21) wh[t] = fch[c * 21 + t];
    else if (t < 42) wv[t - 21] = fcw[c * 21 + (t - 21)];
    __syncthreads();
    const float* x2 = xp + ((long)n * CCH + DH + c) * HW;
    float s = 0.f;
    for (int hw = t; hw < HW; hw += 256) {
        int h = hw / 56, w = hw - h * 56;
        s += x2[hw] + dwconv(x2, wh, h, w, 3, 7) + dwconv(x2, wv, h, w, 7, 3);
    }
    red[t] = s;
    __syncthreads();
    for (int st = 128; st > 0; st >>= 1) {
        if (t < st) red[t] += red[t + st];
        __syncthreads();
    }
    if (t == 0) attmean[nc] = red[0] * (1.0f / HW);
}

// Tiny attention MLP: fc1(40x160)+gelu -> fc2(480x40). Single block.
__global__ __launch_bounds__(512) void att_mlp_k(
    const float* __restrict__ attmean,
    const float* __restrict__ w1, const float* __restrict__ b1,
    const float* __restrict__ w2, const float* __restrict__ b2,
    float* __restrict__ attlin) {
    __shared__ float h1[NB * 40];
    int t = threadIdx.x;
    for (int idx = t; idx < NB * 40; idx += 512) {
        int n = idx / 40, j = idx - n * 40;
        float s = b1[j];
        for (int k = 0; k < DH; ++k) s += attmean[n * DH + k] * w1[j * DH + k];
        h1[idx] = gelu_exact(s);
    }
    __syncthreads();
    for (int idx = t; idx < NB * 480; idx += 512) {
        int n = idx / 480, o = idx - n * 480;
        float s = b2[o];
        for (int k = 0; k < 40; ++k) s += h1[n * 40 + k] * w2[o * 40 + k];
        attlin[idx] = s;
    }
}

// x2_out = xh_path*a0 + xw_path*a1 + x2*a2, softmax over the 3 logits per (n,c).
__global__ __launch_bounds__(256) void combine_k(
    const float* __restrict__ xp,
    const float* __restrict__ fch, const float* __restrict__ fcw,
    const float* __restrict__ attlin, float* __restrict__ x2o) {
    long total = (long)NB * DH * HW;
    for (long idx = (long)blockIdx.x * 256 + threadIdx.x; idx < total;
         idx += (long)gridDim.x * 256) {
        int hw = (int)(idx % HW);
        long nc = idx / HW;
        int c = (int)(nc % DH);
        int n = (int)(nc / DH);
        const float* x2 = xp + ((long)n * CCH + DH + c) * HW;
        int h = hw / 56, w = hw - h * 56;
        float v  = x2[hw];
        float pa = dwconv(x2, fch + c * 21, h, w, 3, 7);   // xw_path
        float pb = dwconv(x2, fcw + c * 21, h, w, 7, 3);   // xh_path
        float l0 = attlin[n * 480 + c * 3 + 0];
        float l1 = attlin[n * 480 + c * 3 + 1];
        float l2 = attlin[n * 480 + c * 3 + 2];
        float mx = fmaxf(l0, fmaxf(l1, l2));
        float e0 = __expf(l0 - mx), e1 = __expf(l1 - mx), e2 = __expf(l2 - mx);
        float inv = 1.0f / (e0 + e1 + e2);
        x2o[idx] = (pb * e0 + pa * e1 + v * e2) * inv;
    }
}

// ---------------------------------------------------------------------------
// Host-side orchestration
// ---------------------------------------------------------------------------
static inline char* carve(char*& p, size_t bytes) {
    char* r = p;
    p += (bytes + 255) & ~(size_t)255;
    return r;
}

extern "C" void kernel_launch(void* const* d_in, const int* in_sizes, int n_in,
                              void* d_out, int out_size, void* d_ws,
                              size_t ws_size, hipStream_t stream) {
    (void)in_sizes; (void)n_in; (void)out_size; (void)ws_size;

    const float* x          = (const float*)d_in[0];
    const float* mlp_pre_w  = (const float*)d_in[1];
    const float* mlp_pre_b  = (const float*)d_in[2];
    const float* bn_s       = (const float*)d_in[3];
    const float* bn_bi      = (const float*)d_in[4];
    const float* bn_m       = (const float*)d_in[5];
    const float* bn_v       = (const float*)d_in[6];
    const float* proj_h_w   = (const float*)d_in[7];
    const float* proj_h_b   = (const float*)d_in[8];
    const float* proj_w_w   = (const float*)d_in[9];
    const float* proj_w_b   = (const float*)d_in[10];
    const float* fuse_h_w   = (const float*)d_in[11];
    const float* fuse_w_w   = (const float*)d_in[12];
    const float* bnh_s      = (const float*)d_in[13];
    const float* bnh_bi     = (const float*)d_in[14];
    const float* bnh_m      = (const float*)d_in[15];
    const float* bnh_v      = (const float*)d_in[16];
    const float* fc_h_w     = (const float*)d_in[17];
    const float* fc_w_w     = (const float*)d_in[18];
    const float* rw_fc1_w   = (const float*)d_in[19];
    const float* rw_fc1_b   = (const float*)d_in[20];
    const float* rw_fc2_w   = (const float*)d_in[21];
    const float* rw_fc2_b   = (const float*)d_in[22];
    const float* fuse_post_w= (const float*)d_in[23];
    const float* emb_h      = (const float*)d_in[24];
    const float* emb_w      = (const float*)d_in[25];
    float* out              = (float*)d_out;

    char* p = (char*)d_ws;
    float*  xp    = (float*)carve(p, (size_t)NB * CCH * HW * 4);
    float*  B0    = (float*)carve(p, (size_t)NB * DH * HW * 4);
    float*  B1    = (float*)carve(p, (size_t)NB * DH * HW * 4);
    float*  posh  = (float*)carve(p, (size_t)DH * HW * 4);
    float*  posw  = (float*)carve(p, (size_t)DH * HW * 4);
    float*  bnpa  = (float*)carve(p, CCH * 4);
    float*  bnpb  = (float*)carve(p, CCH * 4);
    float*  bnha  = (float*)carve(p, DH * 4);
    float*  bnhb  = (float*)carve(p, DH * 4);
    float*  amean = (float*)carve(p, NB * DH * 4);
    float*  alin  = (float*)carve(p, NB * 480 * 4);
    __bf16* Fpre  = (__bf16*)carve(p, (size_t)(CCH/16)*(CCH/32)*512 * 2);
    __bf16* Ffh   = (__bf16*)carve(p, (size_t)(DH/16)*(CCH/32)*512 * 2);
    __bf16* Ffw   = (__bf16*)carve(p, (size_t)(DH/16)*(CCH/32)*512 * 2);
    __bf16* Fpost = (__bf16*)carve(p, (size_t)(CCH/16)*(CCH/32)*512 * 2);

    const long nsX  = (long)CCH * HW;   // 320-channel NCHW stride
    const long nsH  = (long)DH * HW;    // 160-channel NCHW stride

    // --- prep: weight packing, BN folding, positional tables ---
    pack_w_k<<<128, 256, 0, stream>>>(mlp_pre_w,  Fpre,  CCH, CCH);
    pack_w_k<<<128, 256, 0, stream>>>(fuse_h_w,   Ffh,   DH,  CCH);
    pack_w_k<<<128, 256, 0, stream>>>(fuse_w_w,   Ffw,   DH,  CCH);
    pack_w_k<<<128, 256, 0, stream>>>(fuse_post_w,Fpost, CCH, CCH);
    bn_prep_k<<<2, 256, 0, stream>>>(bn_s, bn_bi, bn_m, bn_v, bnpa, bnpb, CCH);
    bn_prep_k<<<1, 256, 0, stream>>>(bnh_s, bnh_bi, bnh_m, bnh_v, bnha, bnhb, DH);
    pos_prep_k<<<(DH * HW + 255) / 256, 256, 0, stream>>>(emb_h, posh);
    pos_prep_k<<<(DH * HW + 255) / 256, 256, 0, stream>>>(emb_w, posw);

    dim3 gemm_blk(256);
    // --- xp = gelu(bn(conv1x1(x, mlp_pre) + b)) ---
    gemm_bf16_k<<<dim3(PTOT / 128, CCH / 32), gemm_blk, 0, stream>>>(
        Fpre, x, nsX, nullptr, 0, CCH, CCH,
        mlp_pre_b, bnpa, bnpb, 1, nullptr, xp, nsX);

    // --- ph = grouped strip conv over (x1 + pos_h) -> B0 ---
    proj_h_kernel<<<dim3(4480, 32), 64, 0, stream>>>(xp, posh, proj_h_w,
                                                     proj_h_b, B0);

    // --- ah = gelu(bn_h(conv1x1([ph, x1], fuse_h))) + pos_w -> B1 ---
    gemm_bf16_k<<<dim3(PTOT / 128, DH / 32), gemm_blk, 0, stream>>>(
        Ffh, B0, nsH, xp, nsX, DH, CCH,
        nullptr, bnha, bnhb, 1, posw, B1, nsH);

    // --- pw = grouped strip conv over ah -> B0 ---
    proj_w_kernel<<<dim3(4480, 32), 64, 0, stream>>>(B1, proj_w_w, proj_w_b, B0);

    // --- x1_out = conv1x1([x1, pw], fuse_w) -> B1 ---
    gemm_bf16_k<<<dim3(PTOT / 128, DH / 32), gemm_blk, 0, stream>>>(
        Ffw, xp, nsX, B0, nsH, DH, CCH,
        nullptr, nullptr, nullptr, 0, nullptr, B1, nsH);

    // --- attention: mean of (paths + x2), MLP, softmax+combine -> B0 ---
    att_reduce_k<<<NB * DH, 256, 0, stream>>>(xp, fc_h_w, fc_w_w, amean);
    att_mlp_k<<<1, 512, 0, stream>>>(amean, rw_fc1_w, rw_fc1_b,
                                     rw_fc2_w, rw_fc2_b, alin);
    combine_k<<<4096, 256, 0, stream>>>(xp, fc_h_w, fc_w_w, alin, B0);

    // --- out = conv1x1([x1_out, x2_out], fuse_post) ---
    gemm_bf16_k<<<dim3(PTOT / 128, CCH / 32), gemm_blk, 0, stream>>>(
        Fpost, B1, nsH, B0, nsH, DH, CCH,
        nullptr, nullptr, nullptr, 0, nullptr, out, nsX);
}
